// BlockDiagonalGRU_721554505852
// MI455X (gfx1250) — matrix operand order repaired and checked by reading, not compiled, not run on hardware
//
#include <hip/hip_runtime.h>
#include <hip/hip_bf16.h>
#include <math.h>

typedef __attribute__((ext_vector_type(16))) _Float16 v16h;
typedef __attribute__((ext_vector_type(8)))  float    v8f;

#define BATCH 1024
#define DIM   4096
#define NB    8
#define BS    512      // block size (hidden & input per block)
#define G3    1536     // 3 * BS
#define BM    128      // batch rows per workgroup
#define BN    32       // hidden cols per workgroup
#define BK    32       // K step
#define LDST  40       // BK + 8 halves padding (row stride = 80B, 16B aligned)

// LDS layout within one stage buffer (offsets in halves)
#define LOFF_X   0                      // 128 * 40 = 5120
#define LOFF_H   5120                   // 128 * 40 = 5120
#define LOFF_WI  10240                  // 3 * 32 * 40 = 3840
#define LOFF_WH  14080                  // 3 * 32 * 40 = 3840
#define STAGE_H  17920                  // halves per stage buffer (35840 B)

// byte offsets of the f16 mirrors inside the single d_ws allocation
#define WSOFF_X   0u
#define WSOFF_H   8388608u              // 1024*4096*2
#define WSOFF_WI  16777216u
#define WSOFF_WH  29360128u             // + 8*1536*512*2

union Frag { v16h v; uint4 q[2]; };

// ---------------------------------------------------------------------------
// fp32 -> f16 conversion pass (feeds the async-DMA GEMM kernel)
// ---------------------------------------------------------------------------
struct alignas(8) H4 { _Float16 v[4]; };

__global__ __launch_bounds__(256)
void cvt_f32_f16(const float* __restrict__ in, _Float16* __restrict__ out, int n4) {
    int i = blockIdx.x * blockDim.x + threadIdx.x;
    if (i < n4) {
        float4 f = ((const float4*)in)[i];
        H4 o;
        o.v[0] = (_Float16)f.x; o.v[1] = (_Float16)f.y;
        o.v[2] = (_Float16)f.z; o.v[3] = (_Float16)f.w;
        ((H4*)out)[i] = o;
    }
}

// ---------------------------------------------------------------------------
// Fused block-diagonal GRU, f16 operands, double-buffered async-to-LDS DMA.
// Async loads use the saddr (GVS) form: one SGPR-pair base (d_ws) + 32-bit
// per-lane offsets -> 7 VGPRs of address state instead of 14.
// ---------------------------------------------------------------------------
__global__ __launch_bounds__(256)
void gru_blockdiag_wmma_async(const _Float16* __restrict__ ws,
                              const float* __restrict__ h,
                              const float* __restrict__ b_ih,
                              const float* __restrict__ b_hh,
                              float* __restrict__ out)
{
    __shared__ _Float16 smem[2][STAGE_H];

    const int tid  = threadIdx.x;
    const int n    = blockIdx.z;
    const int m0   = blockIdx.y * BM;
    const int c0   = blockIdx.x * BN;

    const int wave = tid >> 5;
    const int lane = tid & 31;
    const int lrow = lane & 15;
    const int lhi  = lane >> 4;
    const int wm   = wave * 16;

    v8f acc[6][2];
#pragma unroll
    for (int g = 0; g < 6; ++g)
#pragma unroll
        for (int s = 0; s < 2; ++s)
            acc[g][s] = (v8f)(0.0f);

    // ---- per-thread DMA plan: 7 x 16B segments per stage (28KB / 256 threads)
    // segments: [0,512) X tile, [512,1024) H tile, [1024,1408) Wi, [1408,1792) Wh
    unsigned ldsoff[7];                 // byte offset within a stage buffer
    unsigned voff[7];                   // byte offset from d_ws base (advances 64B/stage)
#pragma unroll
    for (int i = 0; i < 7; ++i) {
        int q = tid + i * 256;
        unsigned lo, go;
        if (q < 512) {                            // X tile: 128 rows x 4 segs
            int row = q >> 2, seg = q & 3;
            lo = (LOFF_X + row * LDST + seg * 8) * 2;
            go = WSOFF_X + ((unsigned)(m0 + row) * DIM + (unsigned)n * BS + seg * 8) * 2;
        } else if (q < 1024) {                    // H tile
            int q2 = q - 512;
            int row = q2 >> 2, seg = q2 & 3;
            lo = (LOFF_H + row * LDST + seg * 8) * 2;
            go = WSOFF_H + ((unsigned)(m0 + row) * DIM + (unsigned)n * BS + seg * 8) * 2;
        } else if (q < 1408) {                    // W_ih: 3 gates x 32 rows x 4 segs
            int q2 = q - 1024;
            int grow = q2 >> 2, seg = q2 & 3;
            int gate = grow >> 5, wrow = grow & 31;
            lo = (LOFF_WI + gate * (32 * LDST) + wrow * LDST + seg * 8) * 2;
            go = WSOFF_WI + ((unsigned)n * G3 * BS
                           + (unsigned)(gate * BS + c0 + wrow) * BS + seg * 8) * 2;
        } else {                                  // W_hh
            int q2 = q - 1408;
            int grow = q2 >> 2, seg = q2 & 3;
            int gate = grow >> 5, wrow = grow & 31;
            lo = (LOFF_WH + gate * (32 * LDST) + wrow * LDST + seg * 8) * 2;
            go = WSOFF_WH + ((unsigned)n * G3 * BS
                           + (unsigned)(gate * BS + c0 + wrow) * BS + seg * 8) * 2;
        }
        ldsoff[i] = lo;
        voff[i]   = go;
    }

    const unsigned long long wsbase = (unsigned long long)(uintptr_t)ws;
    const unsigned ldsbase0 = (unsigned)(uintptr_t)&smem[0][0];
    const unsigned ldsbase1 = (unsigned)(uintptr_t)&smem[1][0];

    auto issue_stage = [&](int buf) {
        unsigned base = buf ? ldsbase1 : ldsbase0;
#pragma unroll
        for (int i = 0; i < 7; ++i) {
            asm volatile("global_load_async_to_lds_b128 %0, %1, %2"
                         :: "v"(base + ldsoff[i]), "v"(voff[i]), "s"(wsbase)
                         : "memory");
            voff[i] += BK * 2;                    // advance 64B to next K tile
        }
    };

    // precomputed fragment row offsets (in halves, invariant across K steps)
    const int aRow  = (wm + lrow) * LDST + lhi * 16;       // A frags
    const int bRow0 = (0 * 16 + lrow) * LDST + lhi * 16;   // B frags, sub-tile 0
    const int bRow1 = (1 * 16 + lrow) * LDST + lhi * 16;   // B frags, sub-tile 1

    // ---- software pipeline: prefetch stage 0, then overlap DMA with WMMA
    issue_stage(0);

    const int NSTEPS = BS / BK;                   // 16
    for (int it = 0; it < NSTEPS; ++it) {
        const int cur = it & 1;
        if (it < NSTEPS - 1) {
            issue_stage(cur ^ 1);                 // fill other buffer while computing
            asm volatile("s_wait_asynccnt 0x7" ::: "memory");  // cur stage landed
        } else {
            asm volatile("s_wait_asynccnt 0x0" ::: "memory");
        }
        __syncthreads();                          // all waves' DMA parts visible

        const _Float16* bp = &smem[cur][0];

        Frag aX, aH;
        {
            const _Float16* p = bp + LOFF_X + aRow;
            aX.q[0] = *(const uint4*)(p);
            aX.q[1] = *(const uint4*)(p + 8);
            const _Float16* p2 = bp + LOFF_H + aRow;
            aH.q[0] = *(const uint4*)(p2);
            aH.q[1] = *(const uint4*)(p2 + 8);
        }

        // ---- 12 WMMAs, B fragments double-buffered (load f+1 while issuing f)
        // fragment id f: g = f>>2, s = (f>>1)&1, w = f&1 (0 = W_ih, 1 = W_hh)
        Frag bf0, bf1;
        auto loadB = [&](Frag& dst, int f) {
            const int g = f >> 2, s = (f >> 1) & 1, w = f & 1;
            const _Float16* p = bp + (w ? LOFF_WH : LOFF_WI) + g * (32 * LDST)
                              + (s ? bRow1 : bRow0);
            dst.q[0] = *(const uint4*)(p);
            dst.q[1] = *(const uint4*)(p + 8);
        };

        loadB(bf0, 0);
#pragma unroll
        for (int f = 0; f < 12; ++f) {
            if (f < 11) loadB((f & 1) ? bf0 : bf1, f + 1);
            const int g = f >> 2, s = (f >> 1) & 1, w = f & 1;
            const Frag& bcur = (f & 1) ? bf1 : bf0;
            const int ai = (w ? 3 : 0) + g;
            acc[ai][s] = __builtin_amdgcn_wmma_f32_16x16x32_f16(
                false, w ? aH.v : aX.v, false, bcur.v, (short)0, acc[ai][s],
                false, false);
        }
        __syncthreads();                          // done reading cur before re-fill
    }

    // ---- fused GRU epilogue
#pragma unroll
    for (int s = 0; s < 2; ++s) {
        const int col  = c0 + s * 16 + lrow;
        const size_t bb = (size_t)n * G3 + col;
        const float bir = b_ih[bb],          bhr = b_hh[bb];
        const float biz = b_ih[bb + BS],     bhz = b_hh[bb + BS];
        const float bin = b_ih[bb + 2 * BS], bhn = b_hh[bb + 2 * BS];
#pragma unroll
        for (int v = 0; v < 8; ++v) {
            const int grow = m0 + wm + v + 8 * lhi;
            const float r = 1.0f / (1.0f + expf(-(acc[0][s][v] + bir + acc[3][s][v] + bhr)));
            const float z = 1.0f / (1.0f + expf(-(acc[1][s][v] + biz + acc[4][s][v] + bhz)));
            const float ng = tanhf(acc[2][s][v] + bin + r * (acc[5][s][v] + bhn));
            const size_t oidx = (size_t)grow * DIM + (size_t)n * BS + col;
            const float hv = h[oidx];
            out[oidx] = ng + z * (hv - ng);
        }
    }
}

// ---------------------------------------------------------------------------
// Fallback: fp32 staging + in-loop cvt, used if ws too small
// ---------------------------------------------------------------------------
__global__ __launch_bounds__(256)
void gru_blockdiag_wmma_direct(const float* __restrict__ x,
                               const float* __restrict__ h,
                               const float* __restrict__ W_ih,
                               const float* __restrict__ W_hh,
                               const float* __restrict__ b_ih,
                               const float* __restrict__ b_hh,
                               float* __restrict__ out)
{
    __shared__ _Float16 sX[BM * LDST];
    __shared__ _Float16 sH[BM * LDST];
    __shared__ _Float16 sWi[3][BN * LDST];
    __shared__ _Float16 sWh[3][BN * LDST];

    const int tid  = threadIdx.x;
    const int n    = blockIdx.z;
    const int m0   = blockIdx.y * BM;
    const int c0   = blockIdx.x * BN;

    const int wave = tid >> 5;
    const int lane = tid & 31;
    const int lrow = lane & 15;
    const int lhi  = lane >> 4;
    const int wm   = wave * 16;

    v8f acc[6][2];
#pragma unroll
    for (int g = 0; g < 6; ++g)
#pragma unroll
        for (int s = 0; s < 2; ++s)
            acc[g][s] = (v8f)(0.0f);

    const int wrow = tid >> 3;
    const int wc4  = (tid & 7) << 2;

    for (int kb = 0; kb < BS; kb += BK) {
#pragma unroll
        for (int i = 0; i < 4; ++i) {
            int q = tid + i * 256;
            int row = q >> 3;
            int c4  = (q & 7) << 2;
            size_t gidx = (size_t)(m0 + row) * DIM + (size_t)n * BS + kb + c4;
            float4 vx = *(const float4*)(x + gidx);
            float4 vh = *(const float4*)(h + gidx);
            _Float16* px = &sX[row * LDST + c4];
            px[0] = (_Float16)vx.x; px[1] = (_Float16)vx.y;
            px[2] = (_Float16)vx.z; px[3] = (_Float16)vx.w;
            _Float16* ph = &sH[row * LDST + c4];
            ph[0] = (_Float16)vh.x; ph[1] = (_Float16)vh.y;
            ph[2] = (_Float16)vh.z; ph[3] = (_Float16)vh.w;
        }
#pragma unroll
        for (int g = 0; g < 3; ++g) {
            size_t widx = (size_t)n * G3 * BS + (size_t)(g * BS + c0 + wrow) * BS + kb + wc4;
            float4 wi = *(const float4*)(W_ih + widx);
            float4 wh = *(const float4*)(W_hh + widx);
            _Float16* pi = &sWi[g][wrow * LDST + wc4];
            pi[0] = (_Float16)wi.x; pi[1] = (_Float16)wi.y;
            pi[2] = (_Float16)wi.z; pi[3] = (_Float16)wi.w;
            _Float16* ph = &sWh[g][wrow * LDST + wc4];
            ph[0] = (_Float16)wh.x; ph[1] = (_Float16)wh.y;
            ph[2] = (_Float16)wh.z; ph[3] = (_Float16)wh.w;
        }
        __syncthreads();

        Frag aX, aH;
        {
            const _Float16* p = &sX[(wm + lrow) * LDST + lhi * 16];
            aX.q[0] = *(const uint4*)(p);
            aX.q[1] = *(const uint4*)(p + 8);
            const _Float16* p2 = &sH[(wm + lrow) * LDST + lhi * 16];
            aH.q[0] = *(const uint4*)(p2);
            aH.q[1] = *(const uint4*)(p2 + 8);
        }

#pragma unroll
        for (int g = 0; g < 3; ++g) {
#pragma unroll
            for (int s = 0; s < 2; ++s) {
                Frag b;
                const _Float16* p = &sWi[g][(s * 16 + lrow) * LDST + lhi * 16];
                b.q[0] = *(const uint4*)(p);
                b.q[1] = *(const uint4*)(p + 8);
                acc[g][s] = __builtin_amdgcn_wmma_f32_16x16x32_f16(
                    false, aX.v, false, b.v, (short)0, acc[g][s], false, false);

                const _Float16* p2 = &sWh[g][(s * 16 + lrow) * LDST + lhi * 16];
                b.q[0] = *(const uint4*)(p2);
                b.q[1] = *(const uint4*)(p2 + 8);
                acc[3 + g][s] = __builtin_amdgcn_wmma_f32_16x16x32_f16(
                    false, aH.v, false, b.v, (short)0, acc[3 + g][s], false, false);
            }
        }
        __syncthreads();
    }

#pragma unroll
    for (int s = 0; s < 2; ++s) {
        const int col  = c0 + s * 16 + lrow;
        const size_t bb = (size_t)n * G3 + col;
        const float bir = b_ih[bb],          bhr = b_hh[bb];
        const float biz = b_ih[bb + BS],     bhz = b_hh[bb + BS];
        const float bin = b_ih[bb + 2 * BS], bhn = b_hh[bb + 2 * BS];
#pragma unroll
        for (int v = 0; v < 8; ++v) {
            const int grow = m0 + wm + v + 8 * lhi;
            const float r = 1.0f / (1.0f + expf(-(acc[0][s][v] + bir + acc[3][s][v] + bhr)));
            const float z = 1.0f / (1.0f + expf(-(acc[1][s][v] + biz + acc[4][s][v] + bhz)));
            const float ng = tanhf(acc[2][s][v] + bin + r * (acc[5][s][v] + bhn));
            const size_t oidx = (size_t)grow * DIM + (size_t)n * BS + col;
            const float hv = h[oidx];
            out[oidx] = ng + z * (hv - ng);
        }
    }
}

// ---------------------------------------------------------------------------
extern "C" void kernel_launch(void* const* d_in, const int* in_sizes, int n_in,
                              void* d_out, int out_size, void* d_ws, size_t ws_size,
                              hipStream_t stream) {
    (void)in_sizes; (void)n_in; (void)out_size;
    const float* x    = (const float*)d_in[0];
    const float* h    = (const float*)d_in[1];
    const float* W_ih = (const float*)d_in[2];
    const float* W_hh = (const float*)d_in[3];
    const float* b_ih = (const float*)d_in[4];
    const float* b_hh = (const float*)d_in[5];
    float* out = (float*)d_out;

    const size_t nXH = (size_t)BATCH * DIM;       // 4,194,304 elements each
    const size_t nW  = (size_t)NB * G3 * BS;      // 6,291,456 elements each
    const size_t need = (nXH * 2 + nW * 2) * sizeof(_Float16);  // 41,943,040 B

    dim3 grid(BS / BN, BATCH / BM, NB);           // 16 x 8 x 8
    dim3 block(256);                              // 8 waves (wave32)

    if (ws_size >= need) {
        _Float16* ws   = (_Float16*)d_ws;
        _Float16* x16  = ws;
        _Float16* h16  = x16 + nXH;
        _Float16* wi16 = h16 + nXH;
        _Float16* wh16 = wi16 + nW;

        cvt_f32_f16<<<(int)(nXH / 4 / 256), 256, 0, stream>>>(x, x16, (int)(nXH / 4));
        cvt_f32_f16<<<(int)(nXH / 4 / 256), 256, 0, stream>>>(h, h16, (int)(nXH / 4));
        cvt_f32_f16<<<(int)(nW / 4 / 256), 256, 0, stream>>>(W_ih, wi16, (int)(nW / 4));
        cvt_f32_f16<<<(int)(nW / 4 / 256), 256, 0, stream>>>(W_hh, wh16, (int)(nW / 4));

        gru_blockdiag_wmma_async<<<grid, block, 0, stream>>>(
            ws, h, b_ih, b_hh, out);
    } else {
        gru_blockdiag_wmma_direct<<<grid, block, 0, stream>>>(
            x, h, W_ih, W_hh, b_ih, b_hh, out);
    }
}
